// GPT2Attention_62371515072661
// MI455X (gfx1250) — matrix operand order repaired and checked by reading, not compiled, not run on hardware
//
#include <hip/hip_runtime.h>
#include <math.h>

typedef __bf16 bf16_t;
typedef __attribute__((ext_vector_type(16))) __bf16 bf16x16;
typedef __attribute__((ext_vector_type(8)))  float  f32x8;

#define BB 4
#define SS 2048
#define DD 768
#define HH 12
#define HD 64

__device__ __forceinline__ bf16_t f2bf(float x) {
  unsigned int u = __builtin_bit_cast(unsigned int, x);
  u += 0x7FFFu + ((u >> 16) & 1u);                 // round-to-nearest-even
  unsigned short s = (unsigned short)(u >> 16);
  return __builtin_bit_cast(bf16_t, s);
}

union FragCast { unsigned int u[8]; bf16x16 v; };

// A fragment: 16x32 bf16, row-major source, ld in elements.
// ISA layout: lanes 0-15 hold M=0..15; VGPR0..3: K=half*8+{0..7}; VGPR4..7: K=16+half*8+{0..7}
__device__ __forceinline__ bf16x16 load_fragA(const bf16_t* base, int ld) {
  const int lane = threadIdx.x & 31;
  const int half = lane >> 4;
  const int r    = lane & 15;
  FragCast t;
#pragma unroll
  for (int j = 0; j < 8; ++j) {
    const int k = (j < 4) ? (half * 8 + 2 * j) : (16 + half * 8 + 2 * (j - 4));
    t.u[j] = *reinterpret_cast<const unsigned int*>(base + (size_t)r * ld + k);
  }
  return t.v;
}

// B fragment: 32x16 bf16. Memory holds B transposed: element B[k][n] at base[n*ld + k].
// ISA layout: lanes 0-15 hold N=0..15; VGPR0..3: K=half*8+{0..7}; VGPR4..7: K=16+half*8+{0..7}
__device__ __forceinline__ bf16x16 load_fragB(const bf16_t* base, int ld) {
  const int lane = threadIdx.x & 31;
  const int half = lane >> 4;
  const int n    = lane & 15;
  FragCast t;
#pragma unroll
  for (int j = 0; j < 8; ++j) {
    const int k = (j < 4) ? (half * 8 + 2 * j) : (16 + half * 8 + 2 * (j - 4));
    t.u[j] = *reinterpret_cast<const unsigned int*>(base + (size_t)n * ld + k);
  }
  return t.v;
}

__device__ __forceinline__ f32x8 wmma_bf16(bf16x16 a, bf16x16 b, f32x8 c) {
  return __builtin_amdgcn_wmma_f32_16x16x32_bf16(false, a, false, b, (short)0, c, false, false);
}

// Register-blocked 32x64 tile per wave: 2 A-frags x 4 B-frags -> 8 WMMAs per k-step.
// A row-major [M,K]; Bt transposed weights [N][K].
__device__ __forceinline__ void gemm_tile_2x4(const bf16_t* __restrict__ A,
                                              const bf16_t* __restrict__ Bt,
                                              int K, int lda, int ldb,
                                              f32x8 acc[2][4]) {
  for (int kk = 0; kk < K; kk += 32) {
    bf16x16 a0 = load_fragA(A + kk, lda);
    bf16x16 a1 = load_fragA(A + (size_t)16 * lda + kk, lda);
    bf16x16 b0 = load_fragB(Bt + kk, ldb);
    bf16x16 b1 = load_fragB(Bt + (size_t)16 * ldb + kk, ldb);
    bf16x16 b2 = load_fragB(Bt + (size_t)32 * ldb + kk, ldb);
    bf16x16 b3 = load_fragB(Bt + (size_t)48 * ldb + kk, ldb);
    acc[0][0] = wmma_bf16(a0, b0, acc[0][0]);
    acc[0][1] = wmma_bf16(a0, b1, acc[0][1]);
    acc[0][2] = wmma_bf16(a0, b2, acc[0][2]);
    acc[0][3] = wmma_bf16(a0, b3, acc[0][3]);
    acc[1][0] = wmma_bf16(a1, b0, acc[1][0]);
    acc[1][1] = wmma_bf16(a1, b1, acc[1][1]);
    acc[1][2] = wmma_bf16(a1, b2, acc[1][2]);
    acc[1][3] = wmma_bf16(a1, b3, acc[1][3]);
  }
}

// ---------------- conversion kernels ----------------
__global__ void cvt_bf16_kernel(const float* __restrict__ x, bf16_t* __restrict__ y, int n) {
  int i = blockIdx.x * blockDim.x + threadIdx.x;
  if (i < n) y[i] = f2bf(x[i]);
}

// w[K][N] f32  ->  wt[N][K] bf16
__global__ void transpose_cvt_kernel(const float* __restrict__ w, bf16_t* __restrict__ wt,
                                     int K, int N) {
  int i = blockIdx.x * blockDim.x + threadIdx.x;
  if (i < K * N) {
    int kk = i / N, n = i % N;
    wt[(size_t)n * K + kk] = f2bf(w[i]);
  }
}

// ---------------- QKV GEMM (32x64 per wave) + split-heads epilogue ----------------
__global__ void __launch_bounds__(256)
gemm_qkv_kernel(const bf16_t* __restrict__ X, const bf16_t* __restrict__ Wt,
                const float* __restrict__ bias,
                bf16_t* __restrict__ Qb, bf16_t* __restrict__ Kb, bf16_t* __restrict__ Vt) {
  const int tiles_n = (3 * DD) / 64;                 // 36
  const int tid = blockIdx.x * 8 + (threadIdx.x >> 5);
  const int m0 = (tid / tiles_n) * 32;
  const int n0 = (tid % tiles_n) * 64;

  f32x8 acc[2][4] = {};
  gemm_tile_2x4(X + (size_t)m0 * DD, Wt + (size_t)n0 * DD, DD, DD, DD, acc);

  const int lane = threadIdx.x & 31;
  const int half = lane >> 4;
#pragma unroll
  for (int ni = 0; ni < 4; ++ni) {
    const int nl   = n0 + ni * 16 + (lane & 15);
    const int part = nl / DD;                        // uniform per 16-col subtile
    const int d    = nl % DD;
    const int h    = d / HD, hd = d % HD;
    const float bv = bias[nl];
#pragma unroll
    for (int mi = 0; mi < 2; ++mi) {
#pragma unroll
      for (int j = 0; j < 8; ++j) {
        const int m = m0 + mi * 16 + j + half * 8;
        const int b = m >> 11, s = m & (SS - 1);
        const bf16_t val = f2bf(acc[mi][ni][j] + bv);
        if (part == 0)      Qb[(((size_t)b * HH + h) * SS + s) * HD + hd] = val;
        else if (part == 1) Kb[(((size_t)b * HH + h) * SS + s) * HD + hd] = val;
        else                Vt[(((size_t)b * HH + h) * HD + hd) * SS + s] = val;
      }
    }
  }
}

// ---------------- flash attention (one wave = 16 query rows) ----------------
__global__ void __launch_bounds__(256)
flash_attn_kernel(const bf16_t* __restrict__ Q, const bf16_t* __restrict__ Km,
                  const bf16_t* __restrict__ Vt, bf16_t* __restrict__ ctx) {
  __shared__ bf16_t pbuf[8][16 * 32];
  const int wib  = threadIdx.x >> 5;
  const int gw   = blockIdx.x * 8 + wib;
  const int lane = threadIdx.x & 31;
  const int half = lane >> 4;
  const int qt = gw & (SS / 16 - 1);                 // 0..127
  const int bh = gw >> 7;                            // 0..47

  const bf16_t* q = Q  + ((size_t)bh * SS + qt * 16) * HD;
  const bf16_t* k = Km + (size_t)bh * SS * HD;
  const bf16_t* v = Vt + (size_t)bh * HD * SS;

  const bf16x16 aq0 = load_fragA(q, HD);
  const bf16x16 aq1 = load_fragA(q + 32, HD);

  f32x8 o0 = {}, o1 = {}, o2 = {}, o3 = {};
  float rm[8], rl[8];
#pragma unroll
  for (int j = 0; j < 8; ++j) { rm[j] = -1e30f; rl[j] = 0.f; }

  bf16_t* P = pbuf[wib];
  const int c = lane & 15;

  for (int kb = 0; kb < SS; kb += 32) {
    // scores: S[16 x 32] = Q(16x64) @ K^T
    f32x8 s0 = {}, s1 = {};
    {
      bf16x16 b00 = load_fragB(k + (size_t)kb * HD, HD);
      bf16x16 b01 = load_fragB(k + (size_t)kb * HD + 32, HD);
      s0 = wmma_bf16(aq0, b00, s0);
      s0 = wmma_bf16(aq1, b01, s0);
      bf16x16 b10 = load_fragB(k + (size_t)(kb + 16) * HD, HD);
      bf16x16 b11 = load_fragB(k + (size_t)(kb + 16) * HD + 32, HD);
      s1 = wmma_bf16(aq0, b10, s1);
      s1 = wmma_bf16(aq1, b11, s1);
    }
    // online softmax: row m = j + half*8 lives in VGPR j, cols across a 16-lane group
#pragma unroll
    for (int j = 0; j < 8; ++j) {
      float a0 = s0[j] * 0.125f;                     // 1/sqrt(64)
      float a1 = s1[j] * 0.125f;
      float tm = fmaxf(a0, a1);
#pragma unroll
      for (int off = 8; off; off >>= 1) tm = fmaxf(tm, __shfl_xor(tm, off));
      const float mnew = fmaxf(rm[j], tm);
      const float p0 = __expf(a0 - mnew);
      const float p1 = __expf(a1 - mnew);
      float ts = p0 + p1;
#pragma unroll
      for (int off = 8; off; off >>= 1) ts += __shfl_xor(ts, off);
      const float alpha = __expf(rm[j] - mnew);
      rm[j] = mnew;
      rl[j] = rl[j] * alpha + ts;
      o0[j] *= alpha; o1[j] *= alpha; o2[j] *= alpha; o3[j] *= alpha;
      const int m = j + half * 8;                    // C-layout -> row-major LDS
      P[m * 32 + c]      = f2bf(p0);
      P[m * 32 + 16 + c] = f2bf(p1);
    }
    __syncthreads();
    const bf16x16 ap = load_fragA(P, 32);            // transpose via LDS bounce
    __syncthreads();
    // O(16x64) += P(16x32) @ V(32x64); Vt is [hd][s] so B-pairs are contiguous
    bf16x16 bv0 = load_fragB(v + kb, SS);
    bf16x16 bv1 = load_fragB(v + (size_t)16 * SS + kb, SS);
    bf16x16 bv2 = load_fragB(v + (size_t)32 * SS + kb, SS);
    bf16x16 bv3 = load_fragB(v + (size_t)48 * SS + kb, SS);
    o0 = wmma_bf16(ap, bv0, o0);
    o1 = wmma_bf16(ap, bv1, o1);
    o2 = wmma_bf16(ap, bv2, o2);
    o3 = wmma_bf16(ap, bv3, o3);
  }

  const int b = bh / HH, h = bh % HH;
#pragma unroll
  for (int j = 0; j < 8; ++j) {
    const int m = j + half * 8;
    const int srow = qt * 16 + m;
    const float inv = 1.0f / rl[j];
    const size_t rowbase = ((size_t)b * SS + srow) * DD + h * HD;
    ctx[rowbase + 0  + c] = f2bf(o0[j] * inv);
    ctx[rowbase + 16 + c] = f2bf(o1[j] * inv);
    ctx[rowbase + 32 + c] = f2bf(o2[j] * inv);
    ctx[rowbase + 48 + c] = f2bf(o3[j] * inv);
  }
}

// ---------------- output projection (32x64 per wave) ----------------
__global__ void __launch_bounds__(256)
gemm_proj_kernel(const bf16_t* __restrict__ Xc, const bf16_t* __restrict__ Wt,
                 const float* __restrict__ bias, float* __restrict__ out) {
  const int tiles_n = DD / 64;                       // 12
  const int tid = blockIdx.x * 8 + (threadIdx.x >> 5);
  const int m0 = (tid / tiles_n) * 32;
  const int n0 = (tid % tiles_n) * 64;

  f32x8 acc[2][4] = {};
  gemm_tile_2x4(Xc + (size_t)m0 * DD, Wt + (size_t)n0 * DD, DD, DD, DD, acc);

  const int lane = threadIdx.x & 31;
  const int half = lane >> 4;
#pragma unroll
  for (int ni = 0; ni < 4; ++ni) {
    const int nl   = n0 + ni * 16 + (lane & 15);
    const float bv = bias[nl];
#pragma unroll
    for (int mi = 0; mi < 2; ++mi) {
#pragma unroll
      for (int j = 0; j < 8; ++j) {
        const int m = m0 + mi * 16 + j + half * 8;
        out[(size_t)m * DD + nl] = acc[mi][ni][j] + bv;
      }
    }
  }
}

extern "C" void kernel_launch(void* const* d_in, const int* in_sizes, int n_in,
                              void* d_out, int out_size, void* d_ws, size_t ws_size,
                              hipStream_t stream) {
  const float* hs     = (const float*)d_in[0];
  const float* W_attn = (const float*)d_in[1];
  const float* b_attn = (const float*)d_in[2];
  const float* W_proj = (const float*)d_in[3];
  const float* b_proj = (const float*)d_in[4];
  float* out = (float*)d_out;

  size_t off = 0;
  auto carve = [&](size_t bytes) {
    void* p = (char*)d_ws + off;
    off += (bytes + 255) & ~(size_t)255;
    return p;
  };
  const size_t nTok = (size_t)BB * SS;               // 8192
  bf16_t* Xb  = (bf16_t*)carve(nTok * DD * 2);       // 12.6 MB
  bf16_t* WaT = (bf16_t*)carve((size_t)3 * DD * DD * 2);
  bf16_t* WpT = (bf16_t*)carve((size_t)DD * DD * 2);
  bf16_t* Qb  = (bf16_t*)carve(nTok * DD * 2);
  bf16_t* Kb  = (bf16_t*)carve(nTok * DD * 2);
  bf16_t* Vt  = (bf16_t*)carve(nTok * DD * 2);
  bf16_t* Cb  = (bf16_t*)carve(nTok * DD * 2);       // total ~68 MB (fits L2)

  { int n = (int)(nTok * DD);
    cvt_bf16_kernel<<<(n + 255) / 256, 256, 0, stream>>>(hs, Xb, n); }
  { int n = DD * 3 * DD;
    transpose_cvt_kernel<<<(n + 255) / 256, 256, 0, stream>>>(W_attn, WaT, DD, 3 * DD); }
  { int n = DD * DD;
    transpose_cvt_kernel<<<(n + 255) / 256, 256, 0, stream>>>(W_proj, WpT, DD, DD); }

  { int waves = (int)(nTok / 32) * (3 * DD / 64);    // 256*36 = 9216, divisible by 8
    gemm_qkv_kernel<<<waves / 8, 256, 0, stream>>>(Xb, WaT, b_attn, Qb, Kb, Vt); }

  { int waves = BB * HH * (SS / 16);                 // 6144, divisible by 8
    flash_attn_kernel<<<waves / 8, 256, 0, stream>>>(Qb, Kb, Vt, Cb); }

  { int waves = (int)(nTok / 32) * (DD / 64);        // 256*12 = 3072, divisible by 8
    gemm_proj_kernel<<<waves / 8, 256, 0, stream>>>(Cb, WpT, b_proj, out); }
}
